// RetNetBlock_73804718014646
// MI455X (gfx1250) — matrix operand order repaired and checked by reading, not compiled, not run on hardware
//
#include <hip/hip_runtime.h>

typedef __attribute__((ext_vector_type(16))) _Float16 v16h;
typedef __attribute__((ext_vector_type(8)))  float    v8f;
typedef __attribute__((ext_vector_type(4)))  unsigned int v4u;
typedef __attribute__((ext_vector_type(8)))  int      v8i;
typedef __attribute__((ext_vector_type(4)))  int      v4i;

#if defined(__has_builtin)
#  if __has_builtin(__builtin_amdgcn_tensor_load_to_lds) && __has_builtin(__builtin_amdgcn_s_wait_tensorcnt)
#    define HAVE_TDM 1
#  else
#    define HAVE_TDM 0
#  endif
#else
#  define HAVE_TDM 0
#endif

union Frag { unsigned int u[8]; v16h h; };

__device__ __forceinline__ v8f wmma16(const Frag& a, const Frag& b, v8f c) {
    return __builtin_amdgcn_wmma_f32_16x16x32_f16(false, a.h, false, b.h,
                                                  (short)0, c, false, false);
}

// A-fragment (16x32 f16): lane L: row = L%16, half = L/16
//   vgpr v<4: K = 2v + 8*half ; v>=4: K = 16 + 2(v-4) + 8*half
__device__ __forceinline__ int kA(int v, int half) {
    return ((v >> 2) << 4) + ((v & 3) << 1) + (half << 3);
}
// B-fragment (32x16 f16): lane L: col = L%16, half = L/16
//   vgpr v<4: K = 2v + 8*half ; v>=4: K = 16 + 2(v-4) + 8*half
__device__ __forceinline__ int kB(int v, int half) {
    return ((v >> 2) << 4) + ((v & 3) << 1) + (half << 3);
}

#if HAVE_TDM
// TDM: load a 2D tile (tile1 rows x tile0 elems of f16) from a row-major tensor
// (row length tdim0, rows tdim1, row stride stride0 elems) into LDS at lds_addr.
// pad_int/pad_amt are the D# pad codes (interval: code c -> 2<<c DWORDs,
// amount: code c -> c+1 DWORDs) giving a padded LDS row stride.
__device__ __forceinline__ void tdm_load_2d(
    unsigned lds_addr, unsigned long long gaddr,
    unsigned tdim0, unsigned tdim1, unsigned tile0, unsigned tile1,
    unsigned stride0, unsigned pad_int, unsigned pad_amt)
{
    v4u g0 = { 1u,                                   // count=1, user mode
               lds_addr,
               (unsigned)gaddr,
               (unsigned)((gaddr >> 32) & 0x01ffffffu) | (2u << 30) }; // type=2
    v8i g1 = { (int)((1u << 16) | (1u << 20) | (pad_int << 22) | (pad_amt << 25)),
               (int)(tdim0 << 16),                    // tensor_dim0[15:0]
               (int)((tdim0 >> 16) | (tdim1 << 16)),  // dim0 hi | dim1 lo
               (int)((tdim1 >> 16) | (tile0 << 16)),  // dim1 hi | tile_dim0
               (int)tile1,                            // tile_dim1 (tile_dim2=0)
               (int)stride0,                          // dim0_stride[31:0]
               0, 0 };
    v4i g2 = {0, 0, 0, 0};
    v4i g3 = {0, 0, 0, 0};
    v8i g4 = {0, 0, 0, 0, 0, 0, 0, 0};
    __builtin_amdgcn_tensor_load_to_lds(g0, g1, g2, g3, g4, 0);
}
#endif

// ---------------- converts ----------------
__global__ void cvt_f32_f16(const float* __restrict__ s, _Float16* __restrict__ d, int n) {
    int i = blockIdx.x * blockDim.x + threadIdx.x;
    if (i < n) d[i] = (_Float16)s[i];
}

__global__ void pack_wqkvg(const float* __restrict__ Wq, const float* __restrict__ Wk,
                           const float* __restrict__ Wv, const float* __restrict__ Wg,
                           _Float16* __restrict__ d) {
    int i = blockIdx.x * blockDim.x + threadIdx.x;          // i over 2048*8192
    if (i >= 2048 * 8192) return;
    int k = i >> 13, n = i & 8191;
    const float* W = (n < 2048) ? Wq : (n < 4096) ? Wk : (n < 6144) ? Wv : Wg;
    int col = n & 2047;
    d[i] = (_Float16)W[(size_t)k * 2048 + col];
}

// ---------------- WMMA GEMM: C(f32,MxN) = A(f16,MxK) * B(f16,KxN) ----------------
__global__ __launch_bounds__(256) void gemm_f16_f32(
    const _Float16* __restrict__ A, const _Float16* __restrict__ B,
    float* __restrict__ C, int M, int N, int K, int lda, int ldb, int ldc) {
    __shared__ _Float16 ldsA[128 * 40];   // [row][k], stride 40 (80B, 16B-aligned)
    __shared__ _Float16 ldsBT[128 * 40];  // [n][k] transposed

    const int t = threadIdx.x, lane = t & 31, wave = t >> 5;
    const int mw = wave & 3, nw = wave >> 2;          // 4 M-waves x 2 N-waves
    const int m0 = blockIdx.y * 128, n0 = blockIdx.x * 128;
    const int m = lane & 15, hf = lane >> 4;
#if HAVE_TDM
    const unsigned ldsA_base = (unsigned)(size_t)(void*)ldsA;
#endif

    v8f acc[2][4];
    for (int i = 0; i < 2; ++i)
        for (int j = 0; j < 4; ++j)
            for (int e = 0; e < 8; ++e) acc[i][j][e] = 0.0f;

    for (int k0 = 0; k0 < K; k0 += 32) {
        __syncthreads();
#if HAVE_TDM
        if (wave == 0) {   // TDM: A tile 128 rows x 32 f16, LDS row stride 40 f16
            unsigned long long ga =
                (unsigned long long)(size_t)(const void*)(A + (size_t)m0 * lda + k0);
            tdm_load_2d(ldsA_base, ga, (unsigned)K, (unsigned)M, 32u, 128u,
                        (unsigned)lda, /*pad 16 DW*/3u, /*+4 DW*/3u);
        }
#else
        {   // stage A: 128x32 f16, 16 f16 per thread
            int r = t >> 1, koff = (t & 1) * 16;
            const uint4* s4 = (const uint4*)(A + (size_t)(m0 + r) * lda + k0 + koff);
            uint4* d4 = (uint4*)(ldsA + r * 40 + koff);
            d4[0] = s4[0]; d4[1] = s4[1];
        }
#endif
        {   // stage B transposed: 32x128 f16 -> [n][k]
            int kr = t >> 3, seg = t & 7;
            const _Float16* src = B + (size_t)(k0 + kr) * ldb + n0 + seg * 16;
            uint4 t0 = ((const uint4*)src)[0];
            uint4 t1 = ((const uint4*)src)[1];
            const _Float16* e0 = (const _Float16*)&t0;
            const _Float16* e1 = (const _Float16*)&t1;
            for (int j = 0; j < 8; ++j) ldsBT[(seg * 16 + j) * 40 + kr] = e0[j];
            for (int j = 0; j < 8; ++j) ldsBT[(seg * 16 + 8 + j) * 40 + kr] = e1[j];
        }
#if HAVE_TDM
        if (wave == 0) __builtin_amdgcn_s_wait_tensorcnt(0);
#endif
        __syncthreads();

        Frag a[2];
        for (int i = 0; i < 2; ++i)
            for (int v = 0; v < 8; ++v)
                a[i].u[v] = *(const unsigned int*)&ldsA[(mw * 32 + i * 16 + m) * 40 + kA(v, hf)];
        for (int j = 0; j < 4; ++j) {
            Frag b;
            for (int v = 0; v < 8; ++v)
                b.u[v] = *(const unsigned int*)&ldsBT[(nw * 64 + j * 16 + m) * 40 + kB(v, hf)];
            acc[0][j] = wmma16(a[0], b, acc[0][j]);
            acc[1][j] = wmma16(a[1], b, acc[1][j]);
        }
    }
    for (int i = 0; i < 2; ++i)
        for (int j = 0; j < 4; ++j)
            for (int r8 = 0; r8 < 8; ++r8) {
                int row = m0 + mw * 32 + i * 16 + r8 + 8 * hf;
                int col = n0 + nw * 64 + j * 16 + m;
                C[(size_t)row * ldc + col] = acc[i][j][r8];
            }
}

// ---------------- xPos rotary on q,k; writes head-major f16 ----------------
__global__ void rope_qk(const float* __restrict__ qkvg, const int* __restrict__ pos,
                        _Float16* __restrict__ qf, _Float16* __restrict__ kf) {
    int idx = blockIdx.x * blockDim.x + threadIdx.x;   // pairs: S * HID/2
    if (idx >= 2048 * 1024) return;
    int s = idx >> 10, rem = idx & 1023;
    int h = rem >> 6, i = rem & 63;
    float p = (float)pos[s];
    float inv = __expf(-((float)(2 * i) * (1.0f / 128.0f)) * 9.2103403720f); // ln(10000)
    float sn, cs;
    __sincosf(p * inv, &sn, &cs);
    float base = ((float)(2 * i) + 51.2f) * (1.0f / 179.2f);     // (2i + 0.4*KD)/(1.4*KD)
    float sc = __expf((p * (1.0f / 512.0f)) * __logf(base));
    size_t o = ((size_t)(h * 2048 + s)) * 128 + 2 * i;

    const float* qp = qkvg + (size_t)s * 8192 + h * 128 + 2 * i;
    float a = qp[0], b = qp[1];
    qf[o]     = (_Float16)((a * cs - b * sn) * sc);
    qf[o + 1] = (_Float16)((b * cs + a * sn) * sc);

    const float* kp = qkvg + (size_t)s * 8192 + 2048 + h * 128 + 2 * i;
    a = kp[0]; b = kp[1];
    float isc = 1.0f / sc;
    kf[o]     = (_Float16)((a * cs - b * sn) * isc);
    kf[o + 1] = (_Float16)((b * cs + a * sn) * isc);
}

__global__ void pack_v(const float* __restrict__ qkvg, _Float16* __restrict__ vf) {
    int idx = blockIdx.x * blockDim.x + threadIdx.x;
    if (idx >= 2048 * 2048) return;
    int s = idx >> 11, c = idx & 2047;
    int h = c >> 7, d = c & 127;
    vf[((size_t)(h * 2048 + s)) * 128 + d] = (_Float16)qkvg[(size_t)s * 8192 + 4096 + c];
}

// ---------------- retention: causal decayed attention, all-WMMA ----------------
__global__ __launch_bounds__(256) void retention_kernel(
    const _Float16* __restrict__ qf, const _Float16* __restrict__ kf,
    const _Float16* __restrict__ vf, float* __restrict__ r) {
#if HAVE_TDM
    __shared__ _Float16 ldsK[64 * 132];   // [key][kd], stride from TDM padding
    #define LDSK_STRIDE 132
#else
    __shared__ _Float16 ldsK[64 * 136];
    #define LDSK_STRIDE 136
#endif
    __shared__ _Float16 ldsVT[128 * 66];  // [vcol][key]
    __shared__ _Float16 ldsS[8 * 16 * 68];// per-wave 16x64 score scratch

    const int qt = blockIdx.x, h = blockIdx.y;
    const int t = threadIdx.x, lane = t & 31, wave = t >> 5;
    const int qi = wave & 3, vh = wave >> 2;      // 4 q-subtiles x 2 v-halves
    const int q0 = qt * 64;
    const int m = lane & 15, hf = lane >> 4;
#if HAVE_TDM
    const unsigned ldsK_base = (unsigned)(size_t)(void*)ldsK;
#endif

    // log gamma for this head: gamma = 1 - exp(linspace(ln(1/32), ln(1/512), 16))
    const float l0 = -3.4657359028f, l1 = -6.2383246250f;
    float gamma = 1.0f - __expf(l0 + (l1 - l0) * ((float)h / 15.0f));
    float lg = __logf(gamma);

    // q fragments straight from global (once per block): rows q0+qi*16+m
    Frag qa[4];
    {
        const _Float16* qrow = qf + ((size_t)(h * 2048 + q0 + qi * 16 + m)) * 128;
        for (int ks = 0; ks < 4; ++ks)
            for (int v = 0; v < 8; ++v)
                qa[ks].u[v] = *(const unsigned int*)&qrow[kA(v, hf) + ks * 32];
    }

    v8f acc[4];
    for (int j = 0; j < 4; ++j)
        for (int e = 0; e < 8; ++e) acc[j][e] = 0.0f;

    for (int kt = 0; kt <= qt; ++kt) {
        __syncthreads();
#if HAVE_TDM
        if (wave == 0) {   // TDM: K tile 64 rows x 128 f16, LDS row stride 132 f16
            unsigned long long ga = (unsigned long long)(size_t)(const void*)
                (kf + ((size_t)(h * 2048 + kt * 64)) * 128);
            tdm_load_2d(ldsK_base, ga, 128u, 2048u, 128u, 64u, 128u,
                        /*pad 64 DW*/5u, /*+2 DW*/1u);
        }
        {   // stage V tile transposed: 64x128 -> [vcol][key]
            int row = t >> 2, seg = t & 3;
            const uint4* sV = (const uint4*)(vf + ((size_t)(h * 2048 + kt * 64 + row)) * 128 + seg * 32);
            uint4 tv[4];
            tv[0] = sV[0]; tv[1] = sV[1]; tv[2] = sV[2]; tv[3] = sV[3];
            const _Float16* ve = (const _Float16*)tv;
            for (int j = 0; j < 32; ++j) ldsVT[(seg * 32 + j) * 66 + row] = ve[j];
        }
        if (wave == 0) __builtin_amdgcn_s_wait_tensorcnt(0);
#else
        {   // stage K tile 64x128 and V tile (transposed) 128x64
            int row = t >> 2, seg = t & 3;
            const uint4* sK = (const uint4*)(kf + ((size_t)(h * 2048 + kt * 64 + row)) * 128 + seg * 32);
            uint4* dK = (uint4*)(ldsK + row * LDSK_STRIDE + seg * 32);
            dK[0] = sK[0]; dK[1] = sK[1]; dK[2] = sK[2]; dK[3] = sK[3];

            const uint4* sV = (const uint4*)(vf + ((size_t)(h * 2048 + kt * 64 + row)) * 128 + seg * 32);
            uint4 tv[4];
            tv[0] = sV[0]; tv[1] = sV[1]; tv[2] = sV[2]; tv[3] = sV[3];
            const _Float16* ve = (const _Float16*)tv;
            for (int j = 0; j < 32; ++j) ldsVT[(seg * 32 + j) * 66 + row] = ve[j];
        }
#endif
        __syncthreads();

        // scores 16x64 = q (16x128) x k^T, decay, -> f16 scratch
        for (int j = 0; j < 4; ++j) {
            v8f s;
            for (int e = 0; e < 8; ++e) s[e] = 0.0f;
            for (int ks = 0; ks < 4; ++ks) {
                Frag kb;
                for (int v = 0; v < 8; ++v)
                    kb.u[v] = *(const unsigned int*)&ldsK[(j * 16 + m) * LDSK_STRIDE + kB(v, hf) + ks * 32];
                s = wmma16(qa[ks], kb, s);
            }
            int kg = kt * 64 + j * 16 + m;                  // this lane's key col
            for (int r8 = 0; r8 < 8; ++r8) {
                int qg = q0 + qi * 16 + r8 + 8 * hf;        // C layout row
                int d = qg - kg;
                float val = (d >= 0) ? s[r8] * __expf(lg * (float)d) : 0.0f;
                ldsS[wave * 16 * 68 + (r8 + 8 * hf) * 68 + j * 16 + m] = (_Float16)val;
            }
        }

        // retained += scores (16x64) x v (64x64-half); K=64 -> 2 steps
        for (int ks2 = 0; ks2 < 2; ++ks2) {
            Frag sa;
            for (int v = 0; v < 8; ++v)
                sa.u[v] = *(const unsigned int*)&ldsS[wave * 16 * 68 + m * 68 + kA(v, hf) + ks2 * 32];
            for (int j2 = 0; j2 < 4; ++j2) {
                Frag vb;
                for (int v = 0; v < 8; ++v)
                    vb.u[v] = *(const unsigned int*)&ldsVT[(vh * 64 + j2 * 16 + m) * 66 + kB(v, hf) + ks2 * 32];
                acc[j2] = wmma16(sa, vb, acc[j2]);
            }
        }
    }

    // write r: [s][h][vhd]
    for (int j2 = 0; j2 < 4; ++j2)
        for (int r8 = 0; r8 < 8; ++r8) {
            int row = q0 + qi * 16 + r8 + 8 * hf;
            int col = vh * 64 + j2 * 16 + m;
            r[((size_t)row * 16 + h) * 128 + col] = acc[j2][r8];
        }
}

// ---------------- groupnorm + silu gate -> y f16 ----------------
__global__ __launch_bounds__(256) void groupnorm_gate(
    const float* __restrict__ r, const float* __restrict__ qkvg,
    const float* __restrict__ gnw, const float* __restrict__ gnb,
    _Float16* __restrict__ y) {
    int t = threadIdx.x, lane = t & 31, wave = t >> 5;
    int row = blockIdx.x * 8 + wave;       // row in [0, S*NH)
    int s = row >> 4, h = row & 15;
    const float* rp = r + (size_t)row * 128;
    float4 v4 = ((const float4*)rp)[lane];
    float e[4] = {v4.x, v4.y, v4.z, v4.w};
    float sum = e[0] + e[1] + e[2] + e[3];
    float sq  = e[0]*e[0] + e[1]*e[1] + e[2]*e[2] + e[3]*e[3];
    for (int off = 16; off; off >>= 1) {
        sum += __shfl_xor(sum, off, 32);
        sq  += __shfl_xor(sq, off, 32);
    }
    float mean = sum * (1.0f / 128.0f);
    float var  = sq * (1.0f / 128.0f) - mean * mean;
    float rinv = rsqrtf(var + 1e-5f);
    for (int i = 0; i < 4; ++i) {
        int d = lane * 4 + i;
        int c = h * 128 + d;
        float val = (e[i] - mean) * rinv * gnw[c] + gnb[c];
        float g = qkvg[(size_t)s * 8192 + 6144 + c];
        float sg = g / (1.0f + __expf(-g));
        y[(size_t)s * 2048 + c] = (_Float16)(sg * val);
    }
}

extern "C" void kernel_launch(void* const* d_in, const int* in_sizes, int n_in,
                              void* d_out, int out_size, void* d_ws, size_t ws_size,
                              hipStream_t stream) {
    const float* x   = (const float*)d_in[0];
    const int*   pos = (const int*)d_in[1];
    const float* Wq  = (const float*)d_in[2];
    const float* Wk  = (const float*)d_in[3];
    const float* Wv  = (const float*)d_in[4];
    const float* Wg  = (const float*)d_in[5];
    const float* Wo  = (const float*)d_in[6];
    const float* gnw = (const float*)d_in[7];
    const float* gnb = (const float*)d_in[8];
    float* out = (float*)d_out;

    char* ws = (char*)d_ws;
    const size_t MB = 1u << 20;
    _Float16* xf16  = (_Float16*)(ws);               //  8 MB
    _Float16* wqkvg = (_Float16*)(ws + 8  * MB);     // 32 MB
    _Float16* wof   = (_Float16*)(ws + 40 * MB);     //  8 MB
    _Float16* y     = (_Float16*)(ws + 48 * MB);     //  8 MB
    _Float16* qf    = (_Float16*)(ws + 56 * MB);     //  8 MB
    _Float16* kf    = (_Float16*)(ws + 64 * MB);     //  8 MB
    _Float16* vf    = (_Float16*)(ws + 72 * MB);     //  8 MB
    float*    rbuf  = (float*)(ws + 80 * MB);        // 16 MB
    float*    qkvg  = (float*)(ws + 96 * MB);        // 64 MB

    const int nx = 2048 * 2048;
    cvt_f32_f16<<<(nx + 255) / 256, 256, 0, stream>>>(x, xf16, nx);
    pack_wqkvg<<<(2048 * 8192 + 255) / 256, 256, 0, stream>>>(Wq, Wk, Wv, Wg, wqkvg);
    cvt_f32_f16<<<(nx + 255) / 256, 256, 0, stream>>>(Wo, wof, nx);

    // qkvg = x @ [Wq|Wk|Wv|Wg]
    gemm_f16_f32<<<dim3(64, 16), 256, 0, stream>>>(xf16, wqkvg, qkvg,
                                                   2048, 8192, 2048, 2048, 8192, 8192);

    rope_qk<<<(2048 * 1024 + 255) / 256, 256, 0, stream>>>(qkvg, pos, qf, kf);
    pack_v<<<(nx + 255) / 256, 256, 0, stream>>>(qkvg, vf);

    retention_kernel<<<dim3(32, 16), 256, 0, stream>>>(qf, kf, vf, rbuf);

    groupnorm_gate<<<(2048 * 16) / 8, 256, 0, stream>>>(rbuf, qkvg, gnw, gnb, y);

    // out = (silu(g)*rn) @ Wo
    gemm_f16_f32<<<dim3(16, 16), 256, 0, stream>>>(y, wof, out,
                                                   2048, 2048, 2048, 2048, 2048, 2048);
    (void)in_sizes; (void)n_in; (void)out_size; (void)ws_size;
}
